// BitMGQA_1726576855417
// MI455X (gfx1250) — 16-bit and 8-bit matrix operand order repaired and checked by reading, compiled, not run on hardware
//
#include <hip/hip_runtime.h>
#include <hip/hip_bf16.h>
#include <stdint.h>

#define EMBED 1024
#define KVDIM 512
#define HEADS 4
#define HD    128
#define BB    4
#define NN    2048
#define TOK   (BB * NN)

typedef __attribute__((ext_vector_type(8)))  int      v8i;
typedef __attribute__((ext_vector_type(8)))  float    v8f;
typedef __attribute__((ext_vector_type(16))) _Float16 v16h;


__device__ __forceinline__ v8i load_a_i8(const int8_t* row0, int ld) {
  const int lane = threadIdx.x & 31;
  const int8_t* p = row0 + (lane & 15) * ld + ((lane >> 4) << 4);
  v8i v;
#pragma unroll
  for (int j = 0; j < 4; ++j) {
    int lo, hi;
    __builtin_memcpy(&lo, p + (j >> 1) * 32 + (j & 1) * 8, 4);
    __builtin_memcpy(&hi, p + (j >> 1) * 32 + (j & 1) * 8 + 4, 4);
    v[2 * j] = lo;
    v[2 * j + 1] = hi;
  }
  return v;
}

__device__ __forceinline__ v8i load_b_i8(const int8_t* col0, int ld) {
  const int lane = threadIdx.x & 31;
  const int8_t* p = col0 + (lane & 15) * ld + ((lane >> 4) << 4);
  v8i v;
#pragma unroll
  for (int j = 0; j < 4; ++j) { int t; __builtin_memcpy(&t, p + j * 4, 4); v[j] = t; }
#pragma unroll
  for (int j = 0; j < 4; ++j) { int t; __builtin_memcpy(&t, p + 32 + j * 4, 4); v[4 + j] = t; }
  return v;
}

__device__ __forceinline__ v16h load_a_f16(const _Float16* row0, int ld) {
  const int lane = threadIdx.x & 31;
  const _Float16* p = row0 + (lane & 15) * ld + ((lane >> 4) << 4);
  v16h v;
#pragma unroll
  for (int i = 0; i < 8; ++i) v[i] = p[i];
#pragma unroll
  for (int i = 0; i < 8; ++i) v[8 + i] = p[8 + i];
  return v;
}

__device__ __forceinline__ v16h load_b_f16(const _Float16* col0, int ld) {
  const int lane = threadIdx.x & 31;
  const _Float16* p = col0 + (lane & 15) * ld + ((lane >> 4) << 4);
  v16h v;
#pragma unroll
  for (int i = 0; i < 16; ++i) v[i] = p[i];
  return v;
}

__global__ __launch_bounds__(256) void wstat_pass1(const float* __restrict__ w, int n,
                                                   float* __restrict__ part) {
  __shared__ float s_sum[256], s_abs[256];
  float s = 0.f, a = 0.f;
  for (int i = blockIdx.x * 256 + threadIdx.x; i < n; i += gridDim.x * 256) {
    float x = w[i];
    s += x;
    a += fabsf(x);
  }
  s_sum[threadIdx.x] = s;
  s_abs[threadIdx.x] = a;
  __syncthreads();
  for (int o = 128; o > 0; o >>= 1) {
    if (threadIdx.x < o) {
      s_sum[threadIdx.x] += s_sum[threadIdx.x + o];
      s_abs[threadIdx.x] += s_abs[threadIdx.x + o];
    }
    __syncthreads();
  }
  if (threadIdx.x == 0) {
    part[blockIdx.x * 2 + 0] = s_sum[0];
    part[blockIdx.x * 2 + 1] = s_abs[0];
  }
}

__global__ __launch_bounds__(256) void wstat_pass2(const float* __restrict__ part, float invn,
                                                   float* __restrict__ stat) {
  __shared__ float s_sum[256], s_abs[256];
  s_sum[threadIdx.x] = part[threadIdx.x * 2 + 0];
  s_abs[threadIdx.x] = part[threadIdx.x * 2 + 1];
  __syncthreads();
  for (int o = 128; o > 0; o >>= 1) {
    if (threadIdx.x < o) {
      s_sum[threadIdx.x] += s_sum[threadIdx.x + o];
      s_abs[threadIdx.x] += s_abs[threadIdx.x + o];
    }
    __syncthreads();
  }
  if (threadIdx.x == 0) {
    stat[0] = s_sum[0] * invn;
    stat[1] = s_abs[0] * invn;
  }
}

__global__ __launch_bounds__(256) void wquant_kernel(const float* __restrict__ w,
                                                     const float* __restrict__ stat,
                                                     int8_t* __restrict__ w8, int n) {
  int i = blockIdx.x * 256 + threadIdx.x;
  if (i >= n) return;
  float d = w[i] - stat[0];
  w8[i] = (int8_t)((d > 0.f) ? 1 : ((d < 0.f) ? -1 : 0));
}

__global__ __launch_bounds__(256) void act_rms_quant(const float* __restrict__ x, int D,
                                                     int8_t* __restrict__ q,
                                                     float* __restrict__ inv) {
  __shared__ float s_a[256], s_b[256];
  const int t = blockIdx.x;
  const float* xr = x + (size_t)t * D;
  float ss = 0.f, am = 0.f;
  for (int i = threadIdx.x; i < D; i += 256) {
    float v = xr[i];
    ss += v * v;
    am = fmaxf(am, fabsf(v));
  }
  s_a[threadIdx.x] = ss;
  s_b[threadIdx.x] = am;
  __syncthreads();
  for (int o = 128; o > 0; o >>= 1) {
    if (threadIdx.x < o) {
      s_a[threadIdx.x] += s_a[threadIdx.x + o];
      s_b[threadIdx.x] = fmaxf(s_b[threadIdx.x], s_b[threadIdx.x + o]);
    }
    __syncthreads();
  }
  const float norm = sqrtf(s_a[0]);
  const float rs = sqrtf((float)D) / fmaxf(norm, 1e-12f);
  const float amax = fmaxf(s_b[0] * rs, 1e-5f);
  const float scale = 127.f / amax;
  for (int i = threadIdx.x; i < D; i += 256) {
    float v = rintf(xr[i] * rs * scale);
    v = fminf(127.f, fmaxf(-128.f, v));
    q[(size_t)t * D + i] = (int8_t)v;
  }
  if (threadIdx.x == 0) inv[t] = amax / 127.f;
}

__global__ __launch_bounds__(256) void gemm_i8_kernel(const int8_t* __restrict__ A,
                                                      const float* __restrict__ invA,
                                                      const int8_t* __restrict__ Bw,
                                                      const float* __restrict__ wstat,
                                                      float* __restrict__ C, int M, int N, int K) {
  const int wave = blockIdx.x * (blockDim.x >> 5) + (threadIdx.x >> 5);
  const int nstrips = N >> 6;
  const int mt = wave / nstrips;
  const int ns = wave - mt * nstrips;
  if (mt * 16 >= M) return;
  const float sw = wstat[1];
  const int8_t* arow = A + (size_t)(mt * 16) * K;
  const int8_t* brow = Bw + (size_t)(ns * 64) * K;
  v8i acc[4] = {};

  v8i a_cur = load_a_i8(arow, K);
  v8i b_cur[4];
#pragma unroll
  for (int t = 0; t < 4; ++t) b_cur[t] = load_b_i8(brow + (size_t)(t * 16) * K, K);

  for (int k = 64; k < K; k += 64) {
    v8i a_nxt = load_a_i8(arow + k, K);
    v8i b_nxt[4];
#pragma unroll
    for (int t = 0; t < 4; ++t) b_nxt[t] = load_b_i8(brow + (size_t)(t * 16) * K + k, K);
#pragma unroll
    for (int t = 0; t < 4; ++t)
      acc[t] = __builtin_amdgcn_wmma_i32_16x16x64_iu8(true, a_cur, true, b_cur[t], acc[t], false,
                                                      false);
    a_cur = a_nxt;
#pragma unroll
    for (int t = 0; t < 4; ++t) b_cur[t] = b_nxt[t];
  }
#pragma unroll
  for (int t = 0; t < 4; ++t)
    acc[t] =
        __builtin_amdgcn_wmma_i32_16x16x64_iu8(true, a_cur, true, b_cur[t], acc[t], false, false);

  const int lane = threadIdx.x & 31;
  const int m0 = mt * 16 + ((lane >> 4) << 3);
#pragma unroll
  for (int r = 0; r < 8; ++r) {
    const int m = m0 + r;
    const float f = sw * invA[m];
#pragma unroll
    for (int t = 0; t < 4; ++t) {
      const int n = ns * 64 + t * 16 + (lane & 15);
      C[(size_t)m * N + n] = (float)acc[t][r] * f;
    }
  }
}

__global__ __launch_bounds__(256) void repack_qkv(const float* __restrict__ qf,
                                                  const float* __restrict__ kf,
                                                  const float* __restrict__ vf,
                                                  _Float16* __restrict__ q16,
                                                  _Float16* __restrict__ k16,
                                                  _Float16* __restrict__ v16t) {
  const int idx = blockIdx.x * 256 + threadIdx.x;
  if (idx >= BB * NN * KVDIM) return;
  const int d = idx & 127;
  const int h = (idx >> 7) & 3;
  const int n = (idx >> 9) & (NN - 1);
  const int b = idx >> 20;
  const size_t tok = (size_t)b * NN + n;
  const size_t bh = (size_t)b * HEADS + h;
  const float qs = qf[tok * EMBED + h * 256 + d] + qf[tok * EMBED + h * 256 + 128 + d];
  q16[(bh * NN + n) * HD + d] = (_Float16)qs;
  k16[(bh * NN + n) * HD + d] = (_Float16)kf[tok * KVDIM + h * HD + d];
  v16t[(bh * HD + d) * NN + n] = (_Float16)vf[tok * KVDIM + h * HD + d];
}

__global__ __launch_bounds__(128) void attention_kernel(const _Float16* __restrict__ Q,
                                                        const _Float16* __restrict__ Km,
                                                        const _Float16* __restrict__ Vt,
                                                        float* __restrict__ O) {
  extern __shared__ char smem[];
  float* S = (float*)smem;
  _Float16* P = (_Float16*)(smem + 16 * NN * sizeof(float));
  const int qb = blockIdx.x & (NN / 16 - 1);
  const int h = (blockIdx.x >> 7) & 3;
  const int b = blockIdx.x >> 9;
  const size_t bh = (size_t)b * HEADS + h;
  const _Float16* Qb = Q + (bh * NN + (size_t)qb * 16) * HD;
  const _Float16* Kb = Km + bh * NN * HD;
  const _Float16* Vb = Vt + bh * HD * NN;
  const int lane = threadIdx.x & 31;
  const int w = threadIdx.x >> 5;

  v16h qa[4];
#pragma unroll
  for (int kc = 0; kc < 4; ++kc) qa[kc] = load_a_f16(Qb + kc * 32, HD);
  for (int sb = w; sb < NN / 16; sb += 4) {
    v16h kb[4];
#pragma unroll
    for (int kc = 0; kc < 4; ++kc)
      kb[kc] = load_b_f16(Kb + (size_t)sb * 16 * HD + kc * 32, HD);
    v8f c = {};
#pragma unroll
    for (int kc = 0; kc < 4; ++kc)
      c = __builtin_amdgcn_wmma_f32_16x16x32_f16(false, qa[kc], false, kb[kc], (short)0, c, false,
                                                 false);
    const int n = sb * 16 + (lane & 15);
    const int m0 = (lane >> 4) << 3;
#pragma unroll
    for (int r = 0; r < 8; ++r) S[(m0 + r) * NN + n] = c[r];
  }
  __syncthreads();

  const float isc = 1.0f / 128.0f;
  for (int r = 4 * w; r < 4 * w + 4; ++r) {
    float mx = -3.0e38f;
    for (int i = lane; i < NN; i += 32) mx = fmaxf(mx, S[r * NN + i]);
#pragma unroll
    for (int o = 16; o > 0; o >>= 1) mx = fmaxf(mx, __shfl_xor(mx, o, 32));
    mx *= isc;
    float sum = 0.f;
    for (int i = lane; i < NN; i += 32) {
      float p = __expf(S[r * NN + i] * isc - mx);
      S[r * NN + i] = p;
      sum += p;
    }
#pragma unroll
    for (int o = 16; o > 0; o >>= 1) sum += __shfl_xor(sum, o, 32);
    const float rinv = 1.f / sum;
    for (int i = lane; i < NN; i += 32) P[r * NN + i] = (_Float16)(S[r * NN + i] * rinv);
  }
  __syncthreads();

  float* Orow = O + ((size_t)b * NN + (size_t)qb * 16) * KVDIM + h * HD;
  for (int dt = w; dt < 8; dt += 4) {
    const _Float16* Vrow = Vb + (size_t)(dt * 16) * NN;
    v8f c = {};
    v16h a_cur = load_a_f16(P, NN);
    v16h b_cur = load_b_f16(Vrow, NN);
    for (int sc = 1; sc < NN / 32; ++sc) {
      v16h a_nxt = load_a_f16(P + sc * 32, NN);
      v16h b_nxt = load_b_f16(Vrow + sc * 32, NN);
      c = __builtin_amdgcn_wmma_f32_16x16x32_f16(false, a_cur, false, b_cur, (short)0, c, false,
                                                 false);
      a_cur = a_nxt;
      b_cur = b_nxt;
    }
    c = __builtin_amdgcn_wmma_f32_16x16x32_f16(false, a_cur, false, b_cur, (short)0, c, false,
                                               false);
    const int n = dt * 16 + (lane & 15);
    const int m0 = (lane >> 4) << 3;
#pragma unroll
    for (int r = 0; r < 8; ++r) Orow[(size_t)(m0 + r) * KVDIM + n] = c[r];
  }
}

__global__ __launch_bounds__(256) void layernorm_kernel(const float* __restrict__ x,
                                                        const float* __restrict__ g,
                                                        const float* __restrict__ bta,
                                                        float* __restrict__ y) {
  __shared__ float s1[256], s2[256];
  const int t = blockIdx.x;
  const float* xr = x + (size_t)t * KVDIM;
  float s = 0.f, sq = 0.f;
  for (int i = threadIdx.x; i < KVDIM; i += 256) {
    float v = xr[i];
    s += v;
    sq += v * v;
  }
  s1[threadIdx.x] = s;
  s2[threadIdx.x] = sq;
  __syncthreads();
  for (int o = 128; o > 0; o >>= 1) {
    if (threadIdx.x < o) {
      s1[threadIdx.x] += s1[threadIdx.x + o];
      s2[threadIdx.x] += s2[threadIdx.x + o];
    }
    __syncthreads();
  }
  const float mu = s1[0] * (1.0f / KVDIM);
  const float var = s2[0] * (1.0f / KVDIM) - mu * mu;
  const float rstd = rsqrtf(var + 1e-5f);
  for (int i = threadIdx.x; i < KVDIM; i += 256)
    y[(size_t)t * KVDIM + i] = (xr[i] - mu) * rstd * g[i] + bta[i];
}

extern "C" void kernel_launch(void* const* d_in, const int* in_sizes, int n_in,
                              void* d_out, int out_size, void* d_ws, size_t ws_size,
                              hipStream_t stream) {
  const float* query = (const float*)d_in[0];
  const float* key_ = (const float*)d_in[1];
  const float* value = (const float*)d_in[2];
  const float* w_q = (const float*)d_in[3];
  const float* w_k = (const float*)d_in[4];
  const float* w_v = (const float*)d_in[5];
  const float* w_o = (const float*)d_in[6];
  const float* ln_g = (const float*)d_in[7];
  const float* ln_b = (const float*)d_in[8];

  uintptr_t cur = (uintptr_t)d_ws;
  auto alloc = [&cur](size_t bytes) -> void* {
    void* p = (void*)cur;
    cur += (bytes + 255) & ~(size_t)255;
    return p;
  };

  float* stats = (float*)alloc(4 * 2 * sizeof(float));
  float* part = (float*)alloc(256 * 2 * sizeof(float));
  int8_t* wq8 = (int8_t*)alloc((size_t)EMBED * EMBED);
  int8_t* wk8 = (int8_t*)alloc((size_t)KVDIM * EMBED);
  int8_t* wv8 = (int8_t*)alloc((size_t)KVDIM * EMBED);
  int8_t* wo8 = (int8_t*)alloc((size_t)EMBED * KVDIM);
  int8_t* xq8 = (int8_t*)alloc((size_t)TOK * EMBED);
  int8_t* xk8 = (int8_t*)alloc((size_t)TOK * EMBED);
  int8_t* xv8 = (int8_t*)alloc((size_t)TOK * EMBED);
  float* invq = (float*)alloc(TOK * sizeof(float));
  float* invk = (float*)alloc(TOK * sizeof(float));
  float* invv = (float*)alloc(TOK * sizeof(float));
  float* qf = (float*)alloc((size_t)TOK * EMBED * sizeof(float));
  float* kf = (float*)alloc((size_t)TOK * KVDIM * sizeof(float));
  float* vf = (float*)alloc((size_t)TOK * KVDIM * sizeof(float));
  _Float16* q16 = (_Float16*)alloc((size_t)TOK * KVDIM * sizeof(_Float16));
  _Float16* k16 = (_Float16*)alloc((size_t)TOK * KVDIM * sizeof(_Float16));
  _Float16* v16t = (_Float16*)alloc((size_t)TOK * KVDIM * sizeof(_Float16));
  float* attn_out = qf;
  float* ln_out = qf + (size_t)TOK * KVDIM;
  int8_t* xo8 = (int8_t*)kf;
  float* invo = (float*)vf;

  const int wsz[4] = {EMBED * EMBED, KVDIM * EMBED, KVDIM * EMBED, EMBED * KVDIM};
  const float* wp[4] = {w_q, w_k, w_v, w_o};
  int8_t* w8p[4] = {wq8, wk8, wv8, wo8};
  for (int i = 0; i < 4; ++i) {
    wstat_pass1<<<256, 256, 0, stream>>>(wp[i], wsz[i], part);
    wstat_pass2<<<1, 256, 0, stream>>>(part, 1.0f / (float)wsz[i], stats + 2 * i);
    wquant_kernel<<<(wsz[i] + 255) / 256, 256, 0, stream>>>(wp[i], stats + 2 * i, w8p[i], wsz[i]);
  }

  act_rms_quant<<<TOK, 256, 0, stream>>>(query, EMBED, xq8, invq);
  act_rms_quant<<<TOK, 256, 0, stream>>>(key_, EMBED, xk8, invk);
  act_rms_quant<<<TOK, 256, 0, stream>>>(value, EMBED, xv8, invv);

  gemm_i8_kernel<<<(TOK / 16) * (EMBED / 64) / 8, 256, 0, stream>>>(xq8, invq, wq8, stats + 0, qf,
                                                                    TOK, EMBED, EMBED);
  gemm_i8_kernel<<<(TOK / 16) * (KVDIM / 64) / 8, 256, 0, stream>>>(xk8, invk, wk8, stats + 2, kf,
                                                                    TOK, KVDIM, EMBED);
  gemm_i8_kernel<<<(TOK / 16) * (KVDIM / 64) / 8, 256, 0, stream>>>(xv8, invv, wv8, stats + 4, vf,
                                                                    TOK, KVDIM, EMBED);

  repack_qkv<<<(BB * NN * KVDIM) / 256, 256, 0, stream>>>(qf, kf, vf, q16, k16, v16t);

  attention_kernel<<<BB * HEADS * (NN / 16), 128,
                     16 * NN * sizeof(float) + 16 * NN * sizeof(_Float16), stream>>>(
      q16, k16, v16t, attn_out);

  layernorm_kernel<<<TOK, 256, 0, stream>>>(attn_out, ln_g, ln_b, ln_out);
  act_rms_quant<<<TOK, 256, 0, stream>>>(ln_out, KVDIM, xo8, invo);
  gemm_i8_kernel<<<(TOK / 16) * (EMBED / 64) / 8, 256, 0, stream>>>(xo8, invo, wo8, stats + 6,
                                                                    (float*)d_out, TOK, EMBED,
                                                                    KVDIM);
}
